// TransformerEncoderLayer_5248450036077
// MI455X (gfx1250) — compile-verified, matrix operand order checked by reading, not run on hardware
//
#include <hip/hip_runtime.h>

typedef __attribute__((ext_vector_type(16))) _Float16 v16h;
typedef __attribute__((ext_vector_type(8)))  _Float16 v8h;
typedef __attribute__((ext_vector_type(8)))  float    v8f;
typedef int v4i __attribute__((vector_size(16)));   // pointee type of async-LDS builtins

#define NTOK 4096
#define DMODEL 1024

#if defined(__has_builtin)
#if __has_builtin(__builtin_amdgcn_global_load_async_to_lds_b128) && \
    __has_builtin(__builtin_amdgcn_s_wait_asynccnt)
#define HAVE_ASYNC_LDS 1
#endif
#endif

// ---- helpers ----------------------------------------------------------------
__device__ __forceinline__ void async_copy16(const _Float16* g, _Float16* l) {
#ifdef HAVE_ASYNC_LDS
  __builtin_amdgcn_global_load_async_to_lds_b128(
      (__attribute__((address_space(1))) v4i*)(void*)g,
      (__attribute__((address_space(3))) v4i*)(void*)l, 0, 0);
#else
  uint4 t = *(const uint4*)g;
  *(uint4*)l = t;
#endif
}
__device__ __forceinline__ void async_wait() {
#ifdef HAVE_ASYNC_LDS
  __builtin_amdgcn_s_wait_asynccnt(0);
#endif
}

// A/B 16-bit fragment (16x32 A or "Bt rows" B): lane holds row (lane&15),
// K chunk base kb = (lane&16)?8:0, elements 0..7 at [kb..kb+7], 8..15 at [kb+16..kb+23].
__device__ __forceinline__ v16h load_frag16(const _Float16* p) {
  v8h lo = *(const v8h*)p;
  v8h hi = *(const v8h*)(p + 16);
  return __builtin_shufflevector(lo, hi, 0,1,2,3,4,5,6,7,8,9,10,11,12,13,14,15);
}

__device__ __forceinline__ v8f wmma_f16(v16h a, v16h b, v8f c) {
  return __builtin_amdgcn_wmma_f32_16x16x32_f16(false, a, false, b, (short)0, c, false, false);
}

// ---- tiled GEMM: out[R x C] = A[R x K] * Bt[C x K]^T (+bias)(+relu)(+residual)
// 128x128 block tile, 8 waves x (32x64), double-buffered LDS, 1 barrier/k-step.
__global__ __launch_bounds__(256) void gemm_f16_wmma(
    const _Float16* __restrict__ A, const _Float16* __restrict__ Bt,
    float* __restrict__ outF, _Float16* __restrict__ outH,
    const float* __restrict__ bias, const float* __restrict__ residual,
    int R, int C, int Kd, int relu)
{
  __shared__ __align__(16) _Float16 As[2][128 * 40];
  __shared__ __align__(16) _Float16 Bs[2][128 * 40];
  const int tid  = threadIdx.x;
  const int lane = tid & 31;
  const int w    = tid >> 5;
  const int grp  = lane >> 4;
  const int ln15 = lane & 15;
  const int kb   = grp ? 8 : 0;
  const int blockRow = blockIdx.y * 128;
  const int blockCol = blockIdx.x * 128;
  const int rowBase  = (w & 3) * 32;
  const int colBase  = (w >> 2) * 64;

  const v8f vzero = {0.f,0.f,0.f,0.f,0.f,0.f,0.f,0.f};
  v8f acc[2][4];
#pragma unroll
  for (int r = 0; r < 2; ++r)
#pragma unroll
    for (int c = 0; c < 4; ++c) acc[r][c] = vzero;

  const int ldRow = tid >> 1;
  const int ldSeg = (tid & 1) * 16;
  const _Float16* gA = A  + (size_t)(blockRow + ldRow) * Kd + ldSeg;
  const _Float16* gB = Bt + (size_t)(blockCol + ldRow) * Kd + ldSeg;
  const int lOff = ldRow * 40 + ldSeg;

  // prologue: stage k=0 tile into buffer 0
#ifdef HAVE_ASYNC_LDS
  async_copy16(gA,     As[0] + lOff);
  async_copy16(gA + 8, As[0] + lOff + 8);
  async_copy16(gB,     Bs[0] + lOff);
  async_copy16(gB + 8, Bs[0] + lOff + 8);
#else
  {
    const uint4* sa = (const uint4*)gA; const uint4* sb = (const uint4*)gB;
    uint4 a0 = sa[0], a1 = sa[1], b0 = sb[0], b1 = sb[1];
    *(uint4*)(As[0] + lOff) = a0; *(uint4*)(As[0] + lOff + 8) = a1;
    *(uint4*)(Bs[0] + lOff) = b0; *(uint4*)(Bs[0] + lOff + 8) = b1;
  }
#endif

  int cur = 0;
  for (int kk = 0; kk < Kd; kk += 32) {
    async_wait();
    __syncthreads();
    const int nxt = cur ^ 1;
    const bool hasNext = (kk + 32) < Kd;
#ifdef HAVE_ASYNC_LDS
    if (hasNext) {
      async_copy16(gA + kk + 32,     As[nxt] + lOff);
      async_copy16(gA + kk + 32 + 8, As[nxt] + lOff + 8);
      async_copy16(gB + kk + 32,     Bs[nxt] + lOff);
      async_copy16(gB + kk + 32 + 8, Bs[nxt] + lOff + 8);
    }
#else
    uint4 a0, a1, b0, b1;
    if (hasNext) {
      const uint4* sa = (const uint4*)(gA + kk + 32);
      const uint4* sb = (const uint4*)(gB + kk + 32);
      a0 = sa[0]; a1 = sa[1]; b0 = sb[0]; b1 = sb[1];
    }
#endif

    const _Float16* Ab = As[cur];
    const _Float16* Bb = Bs[cur];
    v16h af0 = load_frag16(Ab + (rowBase +  0 + ln15) * 40 + kb);
    v16h af1 = load_frag16(Ab + (rowBase + 16 + ln15) * 40 + kb);
#pragma unroll
    for (int c = 0; c < 4; ++c) {
      v16h bf = load_frag16(Bb + (colBase + c * 16 + ln15) * 40 + kb);
      acc[0][c] = wmma_f16(af0, bf, acc[0][c]);
      acc[1][c] = wmma_f16(af1, bf, acc[1][c]);
    }

#ifndef HAVE_ASYNC_LDS
    if (hasNext) {
      *(uint4*)(As[nxt] + lOff) = a0; *(uint4*)(As[nxt] + lOff + 8) = a1;
      *(uint4*)(Bs[nxt] + lOff) = b0; *(uint4*)(Bs[nxt] + lOff + 8) = b1;
    }
#endif
    cur = nxt;
  }

#pragma unroll
  for (int r = 0; r < 2; ++r)
#pragma unroll
    for (int c = 0; c < 4; ++c) {
      int gcol = blockCol + colBase + c * 16 + ln15;
      float bv = bias ? bias[gcol] : 0.f;
#pragma unroll
      for (int vr = 0; vr < 8; ++vr) {
        int grow = blockRow + rowBase + r * 16 + vr + grp * 8;
        float v = acc[r][c][vr] + bv;
        if (relu) v = v > 0.f ? v : 0.f;
        size_t gi = (size_t)grow * C + gcol;
        if (residual) v += residual[gi];
        if (outF) outF[gi] = v;
        if (outH) outH[gi] = (_Float16)v;
      }
    }
}

// ---- fused flash attention: one head per blockIdx.y, 128 queries per block ---
// Double-buffered K/V chunks (32 keys), 1 block barrier per chunk; wave-local
// s_wait_dscnt for the P-reshape staging (LDS ops are in-order per wave).
__global__ __launch_bounds__(256) void attn_fused(
    const _Float16* __restrict__ q16, const _Float16* __restrict__ k16,
    const _Float16* __restrict__ v16, _Float16* __restrict__ o16, int Nn)
{
  __shared__ __align__(16) _Float16 Ks[2][32 * 136];   // keys x dqk
  __shared__ __align__(16) _Float16 Vts[2][128 * 40];  // dv x keys (transposed)
  __shared__ __align__(16) _Float16 Ps[8 * 16 * 40];   // per-wave P staging
  const int tid  = threadIdx.x;
  const int lane = tid & 31;
  const int w    = tid >> 5;
  const int grp  = lane >> 4;
  const int ln15 = lane & 15;
  const int kb   = grp ? 8 : 0;
  const int ho   = blockIdx.y * 128;
  const int qbase = blockIdx.x * 128;
  const int qrow  = qbase + w * 16 + ln15;
  const float invT = 0.03125f; // 1/sqrt(1024)

  v16h aq[4];
#pragma unroll
  for (int kc = 0; kc < 4; ++kc)
    aq[kc] = load_frag16(q16 + (size_t)qrow * DMODEL + ho + kc * 32 + kb);

  const v8f vzero = {0.f,0.f,0.f,0.f,0.f,0.f,0.f,0.f};
  v8f O[8];
  float m8[8], l8[8];
#pragma unroll
  for (int i = 0; i < 8; ++i) { O[i] = vzero; m8[i] = -1e30f; l8[i] = 0.f; }

  // staging geometry: 256 threads move 32 rows x 128 halves per chunk
  const int sr   = tid >> 3;         // key row in chunk (0..31)
  const int sseg = (tid & 7) * 16;   // dqk/dv segment
  const _Float16* gK = k16 + (size_t)sr * DMODEL + ho + sseg;
  const _Float16* gV = v16 + (size_t)sr * DMODEL + ho + sseg;
  const int kOff = sr * 136 + sseg;

  // prologue: stage chunk 0 into buffer 0
#ifdef HAVE_ASYNC_LDS
  async_copy16(gK,     Ks[0] + kOff);
  async_copy16(gK + 8, Ks[0] + kOff + 8);
#else
  {
    const uint4* s = (const uint4*)gK;
    uint4 d0 = s[0], d1 = s[1];
    *(uint4*)(Ks[0] + kOff) = d0; *(uint4*)(Ks[0] + kOff + 8) = d1;
  }
#endif
  {
    const v8h* s = (const v8h*)gV;
    v8h a0 = s[0], a1 = s[1];
#pragma unroll
    for (int e = 0; e < 8; ++e) {
      Vts[0][(sseg + e)     * 40 + sr] = a0[e];
      Vts[0][(sseg + 8 + e) * 40 + sr] = a1[e];
    }
  }

  int cur = 0;
  for (int jb = 0; jb < Nn; jb += 32) {
    async_wait();
    __syncthreads();
    const int nxt = cur ^ 1;
    const bool hasNext = (jb + 32) < Nn;

    v8h va0, va1;
#ifdef HAVE_ASYNC_LDS
    if (hasNext) {
      const _Float16* gk2 = gK + (size_t)(jb + 32) * DMODEL;
      async_copy16(gk2,     Ks[nxt] + kOff);
      async_copy16(gk2 + 8, Ks[nxt] + kOff + 8);
    }
#else
    uint4 rk0, rk1;
    if (hasNext) {
      const uint4* s = (const uint4*)(gK + (size_t)(jb + 32) * DMODEL);
      rk0 = s[0]; rk1 = s[1];
    }
#endif
    if (hasNext) {
      const v8h* s = (const v8h*)(gV + (size_t)(jb + 32) * DMODEL);
      va0 = s[0]; va1 = s[1];
    }

    // ---- scores: S = Q K^T / sqrt(D)
    const _Float16* Kb = Ks[cur];
    v8f s0 = vzero, s1 = vzero;
#pragma unroll
    for (int kc = 0; kc < 4; ++kc) {
      v16h b0 = load_frag16(Kb + ( 0 + ln15) * 136 + kc * 32 + kb);
      v16h b1 = load_frag16(Kb + (16 + ln15) * 136 + kc * 32 + kb);
      s0 = wmma_f16(aq[kc], b0, s0);
      s1 = wmma_f16(aq[kc], b1, s1);
    }

    // ---- online softmax (rows live across 16 lanes of a half-wave)
    float f8[8];
#pragma unroll
    for (int r = 0; r < 8; ++r) {
      float a = s0[r] * invT, b = s1[r] * invT;
      float rm = fmaxf(a, b);
      rm = fmaxf(rm, __shfl_xor(rm, 1, 16));
      rm = fmaxf(rm, __shfl_xor(rm, 2, 16));
      rm = fmaxf(rm, __shfl_xor(rm, 4, 16));
      rm = fmaxf(rm, __shfl_xor(rm, 8, 16));
      float mn = fmaxf(m8[r], rm);
      float fo = __expf(m8[r] - mn);
      float p0 = __expf(a - mn), p1 = __expf(b - mn);
      float rs = p0 + p1;
      rs += __shfl_xor(rs, 1, 16);
      rs += __shfl_xor(rs, 2, 16);
      rs += __shfl_xor(rs, 4, 16);
      rs += __shfl_xor(rs, 8, 16);
      l8[r] = l8[r] * fo + rs;
      m8[r] = mn; f8[r] = fo;
      s0[r] = p0; s1[r] = p1;
    }
#pragma unroll
    for (int dt = 0; dt < 8; ++dt)
#pragma unroll
      for (int r = 0; r < 8; ++r) O[dt][r] *= f8[r];

    // ---- reshape P (D-layout -> A-fragment) via wave-private LDS staging
    _Float16* pw = Ps + w * (16 * 40);
#pragma unroll
    for (int r = 0; r < 8; ++r) {
      int m = r + grp * 8;
      pw[m * 40 + ln15]      = (_Float16)s0[r];
      pw[m * 40 + 16 + ln15] = (_Float16)s1[r];
    }
    asm volatile("s_wait_dscnt 0" ::: "memory"); // wave-local: DS ops in-order per wave
    v16h pa = load_frag16(pw + ln15 * 40 + kb);

    // ---- O += P * V
    const _Float16* Vb = Vts[cur];
#pragma unroll
    for (int dt = 0; dt < 8; ++dt) {
      v16h vb = load_frag16(Vb + (dt * 16 + ln15) * 40 + kb);
      O[dt] = wmma_f16(pa, vb, O[dt]);
    }

    // ---- commit next chunk staging
    if (hasNext) {
#pragma unroll
      for (int e = 0; e < 8; ++e) {
        Vts[nxt][(sseg + e)     * 40 + sr] = va0[e];
        Vts[nxt][(sseg + 8 + e) * 40 + sr] = va1[e];
      }
#ifndef HAVE_ASYNC_LDS
      *(uint4*)(Ks[nxt] + kOff) = rk0; *(uint4*)(Ks[nxt] + kOff + 8) = rk1;
#endif
    }
    cur = nxt;
  }

#pragma unroll
  for (int r = 0; r < 8; ++r) {
    float inv = 1.f / l8[r];
    int q = qbase + w * 16 + r + grp * 8;
#pragma unroll
    for (int dt = 0; dt < 8; ++dt)
      o16[(size_t)q * DMODEL + ho + dt * 16 + ln15] = (_Float16)(O[dt][r] * inv);
  }
}

// ---- row-wise LayerNorm (wave per row), writes f32 + f16 -------------------
__global__ __launch_bounds__(256) void ln_rows(
    const float* __restrict__ y, const float* __restrict__ sc, const float* __restrict__ bi,
    float* __restrict__ x32, _Float16* __restrict__ x16)
{
  int lane = threadIdx.x & 31, w = threadIdx.x >> 5;
  int row = blockIdx.x * 8 + w;
  const float* yr = y + (size_t)row * DMODEL;
  float v[32], s = 0.f, s2 = 0.f;
#pragma unroll
  for (int i = 0; i < 32; ++i) { float t = yr[lane + i * 32]; v[i] = t; s += t; s2 += t * t; }
#pragma unroll
  for (int off = 1; off < 32; off <<= 1) { s += __shfl_xor(s, off, 32); s2 += __shfl_xor(s2, off, 32); }
  float mean = s * (1.f / DMODEL);
  float var  = s2 * (1.f / DMODEL) - mean * mean;
  float inv  = rsqrtf(var + 1e-5f);
#pragma unroll
  for (int i = 0; i < 32; ++i) {
    int c = lane + i * 32;
    float o = (v[i] - mean) * inv * sc[c] + bi[c];
    x32[(size_t)row * DMODEL + c] = o;
    x16[(size_t)row * DMODEL + c] = (_Float16)o;
  }
}

__global__ __launch_bounds__(256) void row_mean(const float* __restrict__ x, float* __restrict__ out) {
  int lane = threadIdx.x & 31, w = threadIdx.x >> 5;
  int row = blockIdx.x * 8 + w;
  const float* xr = x + (size_t)row * DMODEL;
  float s = 0.f;
#pragma unroll
  for (int i = 0; i < 32; ++i) s += xr[lane + i * 32];
#pragma unroll
  for (int off = 1; off < 32; off <<= 1) s += __shfl_xor(s, off, 32);
  if (lane == 0) out[row] = s * (1.f / DMODEL);
}

__global__ __launch_bounds__(256) void init_x(const float* __restrict__ xin,
                                              float* __restrict__ x32, _Float16* __restrict__ x16, int n) {
  int i = blockIdx.x * 256 + threadIdx.x;
  if (i < n) { float v = xin[i]; x32[i] = v; x16[i] = (_Float16)v; }
}

// mode 0: out[r][p] = W[p][r]                        (plain transpose, W1/W2)
// mode 1: out[h*128+c][p] = W[p][c*8+h]              (QKV: de-interleave heads)
// mode 2: out[col][h*128+d] = W[d*8+h][col]          (Wo)
__global__ __launch_bounds__(256) void prep_w(const float* __restrict__ W, _Float16* __restrict__ out, int mode) {
  int idx = blockIdx.x * 256 + threadIdx.x;
  int r = idx >> 10, p = idx & 1023;
  int src;
  if (mode == 0)      src = p * 1024 + r;
  else if (mode == 1) src = p * 1024 + ((r & 127) * 8 + (r >> 7));
  else                src = (((p & 127) * 8) + (p >> 7)) * 1024 + r;
  out[idx] = (_Float16)W[src];
}

extern "C" void kernel_launch(void* const* d_in, const int* in_sizes, int n_in,
                              void* d_out, int out_size, void* d_ws, size_t ws_size,
                              hipStream_t stream) {
  (void)in_sizes; (void)n_in; (void)out_size; (void)ws_size;
  const float* x_in = (const float*)d_in[0];
  char* ws = (char*)d_ws;
  float*    x32  = (float*)(ws + (size_t)0);
  _Float16* x16  = (_Float16*)(ws + ((size_t)16 << 20));
  _Float16* w16  = (_Float16*)(ws + ((size_t)24 << 20));
  _Float16* q16  = (_Float16*)(ws + ((size_t)36 << 20));
  _Float16* k16  = (_Float16*)(ws + ((size_t)44 << 20));
  _Float16* v16  = (_Float16*)(ws + ((size_t)52 << 20));
  _Float16* ah16 = (_Float16*)(ws + ((size_t)60 << 20)); // attn-out / FFN-hidden
  float*    y32  = (float*)(ws + ((size_t)68 << 20));

  const size_t WSLOT = 1024 * 1024;
  _Float16 *wq = w16, *wk = w16 + WSLOT, *wv = w16 + 2 * WSLOT,
           *wo = w16 + 3 * WSLOT, *w1 = w16 + 4 * WSLOT, *w2 = w16 + 5 * WSLOT;

  dim3 blk(256);
  dim3 gGemm(DMODEL / 128, NTOK / 128); // (8, 32)
  dim3 gAttn(NTOK / 128, 8);            // (32, 8)

  init_x<<<(NTOK * DMODEL + 255) / 256, blk, 0, stream>>>(x_in, x32, x16, NTOK * DMODEL);

  for (int L = 0; L < 3; ++L) {
    int b = 1 + 12 * L;
    const float* Wq = (const float*)d_in[b + 0];
    const float* Wk = (const float*)d_in[b + 1];
    const float* Wv = (const float*)d_in[b + 2];
    const float* Wo = (const float*)d_in[b + 3];
    const float* l1s = (const float*)d_in[b + 4];
    const float* l1b = (const float*)d_in[b + 5];
    const float* W1 = (const float*)d_in[b + 6];
    const float* b1 = (const float*)d_in[b + 7];
    const float* W2 = (const float*)d_in[b + 8];
    const float* b2 = (const float*)d_in[b + 9];
    const float* l2s = (const float*)d_in[b + 10];
    const float* l2b = (const float*)d_in[b + 11];

    prep_w<<<4096, blk, 0, stream>>>(Wq, wq, 1);
    prep_w<<<4096, blk, 0, stream>>>(Wk, wk, 1);
    prep_w<<<4096, blk, 0, stream>>>(Wv, wv, 1);
    prep_w<<<4096, blk, 0, stream>>>(Wo, wo, 2);
    prep_w<<<4096, blk, 0, stream>>>(W1, w1, 0);
    prep_w<<<4096, blk, 0, stream>>>(W2, w2, 0);

    gemm_f16_wmma<<<gGemm, blk, 0, stream>>>(x16, wq, nullptr, q16, nullptr, nullptr, NTOK, DMODEL, DMODEL, 0);
    gemm_f16_wmma<<<gGemm, blk, 0, stream>>>(x16, wk, nullptr, k16, nullptr, nullptr, NTOK, DMODEL, DMODEL, 0);
    gemm_f16_wmma<<<gGemm, blk, 0, stream>>>(x16, wv, nullptr, v16, nullptr, nullptr, NTOK, DMODEL, DMODEL, 0);

    attn_fused<<<gAttn, blk, 0, stream>>>(q16, k16, v16, ah16, NTOK);

    gemm_f16_wmma<<<gGemm, blk, 0, stream>>>(ah16, wo, y32, nullptr, nullptr, x32, NTOK, DMODEL, DMODEL, 0);
    ln_rows<<<NTOK / 8, blk, 0, stream>>>(y32, l1s, l1b, x32, x16);

    gemm_f16_wmma<<<gGemm, blk, 0, stream>>>(x16, w1, nullptr, ah16, b1, nullptr, NTOK, DMODEL, DMODEL, 1);
    gemm_f16_wmma<<<gGemm, blk, 0, stream>>>(ah16, w2, y32, nullptr, b2, x32, NTOK, DMODEL, DMODEL, 0);
    ln_rows<<<NTOK / 8, blk, 0, stream>>>(y32, l2s, l2b, x32, x16);
  }

  row_mean<<<NTOK / 8, blk, 0, stream>>>(x32, (float*)d_out);
}
